// RetNet_70720931496218
// MI455X (gfx1250) — matrix operand order repaired and checked by reading, not compiled, not run on hardware
//
#include <hip/hip_runtime.h>
#include <hip/hip_bf16.h>
#include <math.h>

// ---------------- problem constants (from reference) ----------------
#define LAYERS 2
#define BB     4
#define SS     2048
#define DD     256
#define HH     8
#define HDIM   32            // DD / HH
#define FFND   1024          // FFN * DD
#define NTOK   (BB * SS)     // 8192 rows
#define EPSF   1e-5f

typedef __attribute__((ext_vector_type(16))) __bf16 v16bf;
typedef __attribute__((ext_vector_type(8)))  __bf16 v8bf;
typedef __attribute__((ext_vector_type(8)))  float  v8f;

static __device__ __forceinline__ __bf16 to_bf(float x) { return (__bf16)x; }

static __device__ __forceinline__ v16bf cat16(v8bf lo, v8bf hi) {
  return __builtin_shufflevector(lo, hi, 0, 1, 2, 3, 4, 5, 6, 7,
                                         8, 9, 10, 11, 12, 13, 14, 15);
}

// Async 16B global -> LDS copy (ASYNCcnt-tracked, bypasses VGPRs).
// Low 32 bits of a generic LDS pointer are the LDS byte offset.
static __device__ __forceinline__ void async_copy_b128(void* lds_dst,
                                                       const void* gsrc) {
  const unsigned ldsoff = (unsigned)(uintptr_t)lds_dst;
  asm volatile("global_load_async_to_lds_b128 %0, %1, off"
               :: "v"(ldsoff), "v"(gsrc) : "memory");
}
static __device__ __forceinline__ void async_wait0() {
  asm volatile("s_wait_asynccnt 0x0" ::: "memory");
}

// =====================================================================
// LayerNorm over D=256: one block (256 threads) per row.
// =====================================================================
__global__ __launch_bounds__(256) void ln_kernel(const float* __restrict__ x,
                                                 const float* __restrict__ w,
                                                 const float* __restrict__ b,
                                                 float* __restrict__ out) {
  __shared__ float red[256];
  const int r = blockIdx.x;
  const int t = threadIdx.x;
  const float v = x[(size_t)r * DD + t];
  red[t] = v;
  __syncthreads();
  for (int s = 128; s > 0; s >>= 1) { if (t < s) red[t] += red[t + s]; __syncthreads(); }
  const float mu = red[0] * (1.0f / DD);
  __syncthreads();
  const float d = v - mu;
  red[t] = d * d;
  __syncthreads();
  for (int s = 128; s > 0; s >>= 1) { if (t < s) red[t] += red[t + s]; __syncthreads(); }
  const float rsig = rsqrtf(red[0] * (1.0f / DD) + EPSF);
  out[(size_t)r * DD + t] = d * rsig * w[t] + b[t];
}

// =====================================================================
// Weight pack: WT[n][k] = bf16(W[k][n]).  64x64 tiles through LDS.
// =====================================================================
__global__ __launch_bounds__(256) void wpack_kernel(const float* __restrict__ W,
                                                    __bf16* __restrict__ WT,
                                                    int K, int N) {
  __shared__ alignas(16) __bf16 t[64][72];     // 144B row stride (16B mult)
  const int tid = threadIdx.x;
  const int k0 = blockIdx.x * 64;
  const int n0 = blockIdx.y * 64;
#pragma unroll
  for (int i = 0; i < 4; ++i) {
    const int f = tid + i * 256;
    const int r = f >> 4, c4 = f & 15;
    const float4 w4 = *(const float4*)&W[(size_t)(k0 + r) * N + n0 + c4 * 4];
    __bf16* dst = &t[r][c4 * 4];
    dst[0] = to_bf(w4.x); dst[1] = to_bf(w4.y); dst[2] = to_bf(w4.z); dst[3] = to_bf(w4.w);
  }
  __syncthreads();
  const int n = tid >> 2, ch = tid & 3;
  v8bf o0, o1;
#pragma unroll
  for (int j = 0; j < 8; ++j) {
    o0[j] = t[ch * 16 + j][n];
    o1[j] = t[ch * 16 + 8 + j][n];
  }
  __bf16* dst = &WT[(size_t)(n0 + n) * K + k0 + ch * 16];
  *(v8bf*)(dst)     = o0;
  *(v8bf*)(dst + 8) = o1;
}

// =====================================================================
// V pack: VT[b][h][hd][s] = bf16(V[b][s][h*HD+hd]).  64s x 32hd tiles.
// =====================================================================
__global__ __launch_bounds__(256) void vtpack_kernel(const float* __restrict__ V,
                                                     __bf16* __restrict__ VT) {
  __shared__ alignas(16) __bf16 t[64][40];
  const int tid = threadIdx.x;
  const int s0  = (blockIdx.x & (SS / 64 - 1)) * 64;
  const int bh  = blockIdx.x / (SS / 64);
  const int h   = bh & (HH - 1), b = bh / HH;
#pragma unroll
  for (int i = 0; i < 2; ++i) {
    const int f = tid + i * 256;
    const int s = f >> 3, c4 = f & 7;
    const float4 v4 = *(const float4*)&V[(size_t)(b * SS + s0 + s) * DD + h * HDIM + c4 * 4];
    __bf16* dst = &t[s][c4 * 4];
    dst[0] = to_bf(v4.x); dst[1] = to_bf(v4.y); dst[2] = to_bf(v4.z); dst[3] = to_bf(v4.w);
  }
  __syncthreads();
  const int hd = tid >> 3, ch = tid & 7;
  v8bf o;
#pragma unroll
  for (int j = 0; j < 8; ++j) o[j] = t[ch * 8 + j][hd];
  *(v8bf*)&VT[((size_t)(b * HH + h) * HDIM + hd) * SS + s0 + ch * 8] = o;
}

// =====================================================================
// xPos rotary: reads fp32 q/k, writes rotated+scaled bf16 packed q/k.
// =====================================================================
__global__ __launch_bounds__(256) void xpos_kernel(const float* __restrict__ q,
                                                   const float* __restrict__ k,
                                                   __bf16* __restrict__ qbf,
                                                   __bf16* __restrict__ kbf) {
  const int idx = blockIdx.x * blockDim.x + threadIdx.x;   // NTOK * DD/2 threads
  const int tok = idx >> 7;
  const int p   = idx & 127;
  const int pos = tok & (SS - 1);
  const int h   = p >> 4;
  const int j   = p & 15;
  const int base = h * HDIM + 2 * j;

  const float sv      = (2.0f * j + 0.4f * HDIM) / (1.4f * HDIM);
  const float scale   = powf(sv, (float)pos / 512.0f);
  const float invfreq = powf(10000.0f, -(float)j / 16.0f);
  const float ang = (float)pos * invfreq;
  float sn, cs; __sincosf(ang, &sn, &cs);

  const size_t o = (size_t)tok * DD + base;
  { float e0 = q[o], e1 = q[o + 1];
    union { __bf16 h2[2]; unsigned u; } pq;
    pq.h2[0] = to_bf((e0 * cs - e1 * sn) * scale);
    pq.h2[1] = to_bf((e1 * cs + e0 * sn) * scale);
    *(unsigned*)&qbf[o] = pq.u; }
  { float iscale = 1.0f / scale;
    float e0 = k[o], e1 = k[o + 1];
    union { __bf16 h2[2]; unsigned u; } pk;
    pk.h2[0] = to_bf((e0 * cs - e1 * sn) * iscale);
    pk.h2[1] = to_bf((e1 * cs + e0 * sn) * iscale);
    *(unsigned*)&kbf[o] = pk.u; }
}

// =====================================================================
// Elementwise gate: t = silu(g) * y   (float4 vectorized)
// =====================================================================
__global__ __launch_bounds__(256) void gate_kernel(const float4* __restrict__ g,
                                                   const float4* __restrict__ y,
                                                   float4* __restrict__ t) {
  const size_t i = (size_t)blockIdx.x * blockDim.x + threadIdx.x;
  const float4 gv = g[i];
  const float4 yv = y[i];
  float4 o;
  o.x = gv.x * (1.0f / (1.0f + expf(-gv.x))) * yv.x;
  o.y = gv.y * (1.0f / (1.0f + expf(-gv.y))) * yv.y;
  o.z = gv.z * (1.0f / (1.0f + expf(-gv.z))) * yv.z;
  o.w = gv.w * (1.0f / (1.0f + expf(-gv.w))) * yv.w;
  t[i] = o;
}

// =====================================================================
// bf16-WMMA GEMM: Out[M,N] = A[M,K] @ B[K,N] (+bias)(gelu)(+resid)
// A: fp32 row-major (converted on stage).  BT: pre-packed bf16 [n][k],
// staged with ASYNC global->LDS b128 copies (no VGPR round-trip).
// Block tile 64x64, 8 waves (4 x 2), wave tile 16x32 (2 WMMAs / K-step).
// flags: 1=bias, 2=gelu(exact), 4=add resid
// =====================================================================
__global__ __launch_bounds__(256) void gemm_bf16_kernel(
    const float* __restrict__ A, const __bf16* __restrict__ BT,
    float* __restrict__ Out, int M, int N, int K,
    const float* __restrict__ bias, const float* __restrict__ resid, int flags) {
  __shared__ alignas(16) __bf16 As[64][40];    // [m][k], 80B row stride
  __shared__ alignas(16) __bf16 BsT[64][40];   // [n][k]
  const int tid  = threadIdx.x;
  const int lane = tid & 31;
  const int w    = tid >> 5;
  const int wm   = w >> 1;
  const int wn   = w & 1;
  const int m0   = blockIdx.x * 64;
  const int n0   = blockIdx.y * 64;
  const int r16  = lane & 15;
  const int hb   = lane >> 4;

  v8f acc0 = {}, acc1 = {};

  for (int k0 = 0; k0 < K; k0 += 32) {
    // ---- BT tile 64n x 32k bf16: async global->LDS, 1 x b128 / thread ----
    {
      const int n  = tid >> 2, ch = tid & 3;
      async_copy_b128(&BsT[n][ch * 8], &BT[(size_t)(n0 + n) * K + k0 + ch * 8]);
    }
    // ---- A tile 64x32 fp32 -> bf16 (2 float4 / thread), overlaps async ----
#pragma unroll
    for (int i = 0; i < 2; ++i) {
      const int f  = tid + i * 256;
      const int r  = f >> 3, c4 = f & 7;
      const float4 a4 = *(const float4*)&A[(size_t)(m0 + r) * K + k0 + c4 * 4];
      __bf16* dst = &As[r][c4 * 4];
      dst[0] = to_bf(a4.x); dst[1] = to_bf(a4.y);
      dst[2] = to_bf(a4.z); dst[3] = to_bf(a4.w);
    }
    if (k0 + 32 < K) {   // prefetch next A tile (global_prefetch_b8)
      __builtin_prefetch(&A[(size_t)(m0 + (tid >> 3)) * K + k0 + 32 + (tid & 7) * 4], 0, 0);
    }
    async_wait0();       // drain this wave's async LDS writes
    __syncthreads();

    const v8bf alo = *(const v8bf*)&As[wm * 16 + r16][hb * 16];
    const v8bf ahi = *(const v8bf*)&As[wm * 16 + r16][hb * 16 + 8];
    const v16bf af = cat16(alo, ahi);
    const v8bf b0l = *(const v8bf*)&BsT[wn * 32 + r16][hb * 16];
    const v8bf b0h = *(const v8bf*)&BsT[wn * 32 + r16][hb * 16 + 8];
    const v8bf b1l = *(const v8bf*)&BsT[wn * 32 + 16 + r16][hb * 16];
    const v8bf b1h = *(const v8bf*)&BsT[wn * 32 + 16 + r16][hb * 16 + 8];

    acc0 = __builtin_amdgcn_wmma_f32_16x16x32_bf16(false, af, false, cat16(b0l, b0h),
                                                   (short)0, acc0, false, false);
    acc1 = __builtin_amdgcn_wmma_f32_16x16x32_bf16(false, af, false, cat16(b1l, b1h),
                                                   (short)0, acc1, false, false);
    __syncthreads();
  }

  const int rbase = m0 + wm * 16 + hb * 8;
  const int cg0   = n0 + wn * 32 + r16;
  const int cg1   = cg0 + 16;
#pragma unroll
  for (int i = 0; i < 8; ++i) {
    const int rowg = rbase + i;
    float v0 = acc0[i], v1 = acc1[i];
    if (flags & 1) { v0 += bias[cg0]; v1 += bias[cg1]; }
    if (flags & 2) {
      v0 = 0.5f * v0 * (1.0f + erff(v0 * 0.70710678118f));
      v1 = 0.5f * v1 * (1.0f + erff(v1 * 0.70710678118f));
    }
    if (flags & 4) {
      v0 += resid[(size_t)rowg * N + cg0];
      v1 += resid[(size_t)rowg * N + cg1];
    }
    Out[(size_t)rowg * N + cg0] = v0;
    Out[(size_t)rowg * N + cg1] = v1;
  }
}

// =====================================================================
// Retention attention, flash-style: one wave per (b, h, 16-query tile).
// All operands pre-packed bf16: q/k row-major (B,S,D), v transposed
// (B,H,HD,S).  Every fragment gather is a 32B bf16 b128 pair, zero
// conversions in the hot loop.  OOB keys decay-mask to zero.
// =====================================================================
__global__ __launch_bounds__(256) void retention_kernel(
    const __bf16* __restrict__ Qb, const __bf16* __restrict__ Kb,
    const __bf16* __restrict__ VT,
    const float* __restrict__ gnw, const float* __restrict__ gnb,
    float* __restrict__ Y) {
  __shared__ alignas(16) __bf16 attb[8][16 * 32];   // per-wave masked scores
  __shared__ alignas(16) float  ystb[8][16 * 32];   // per-wave y staging
  const int tid  = threadIdx.x;
  const int lane = tid & 31;
  const int wv   = tid >> 5;
  const int task = blockIdx.x * 8 + wv;             // B*H*(S/16) tasks
  const int n0   = (task & (SS / 16 - 1)) * 16;
  const int bh   = task / (SS / 16);
  const int h    = bh & (HH - 1);
  const int b    = bh / HH;

  __bf16* att = attb[wv];
  float*  yst = ystb[wv];

  const float lg    = -3.4657359028f + (float)h * (-6.2383246250f + 3.4657359028f) / (HH - 1);
  const float gdec  = 1.0f - expf(lg);
  const float log2g = log2f(gdec);

  const __bf16* qb  = Qb + (size_t)b * SS * DD + h * HDIM;
  const __bf16* kb  = Kb + (size_t)b * SS * DD + h * HDIM;
  const __bf16* vtb = VT + (size_t)(b * HH + h) * HDIM * SS;

  const int r16 = lane & 15;
  const int hb  = lane >> 4;
  const int rowoff = hb * 8;

  // q A-fragment: two bf16 b128 loads
  const __bf16* qr = qb + (size_t)(n0 + r16) * DD + hb * 16;
  const v16bf aq = cat16(*(const v8bf*)qr, *(const v8bf*)(qr + 8));

  v8f c0 = {}, c1 = {};
  for (int m0 = 0; m0 <= n0 + 15; m0 += 32) {
#pragma unroll
    for (int sub = 0; sub < 2; ++sub) {
      const int mbase = m0 + sub * 16;
      const __bf16* kr = kb + (size_t)(mbase + r16) * DD + hb * 16;
      const v16bf bk = cat16(*(const v8bf*)kr, *(const v8bf*)(kr + 8));
      v8f sc = {};
      sc = __builtin_amdgcn_wmma_f32_16x16x32_bf16(false, aq, false, bk,
                                                   (short)0, sc, false, false);
      const int mcol = mbase + r16;
#pragma unroll
      for (int i = 0; i < 8; ++i) {
        const int nq = n0 + i + rowoff;
        const float dm = (nq >= mcol && mcol < SS)
                             ? exp2f((float)(nq - mcol) * log2g) : 0.0f;
        att[(i + rowoff) * 32 + r16 + sub * 16] = to_bf(sc[i] * dm);
      }
    }
    // reload scores as A fragment (DS in-order within a wave)
    const v16bf aa = cat16(*(const v8bf*)&att[r16 * 32 + hb * 16],
                           *(const v8bf*)&att[r16 * 32 + hb * 16 + 8]);
#pragma unroll
    for (int half = 0; half < 2; ++half) {
      const __bf16* vr = vtb + (size_t)(half * 16 + r16) * SS + m0 + hb * 16;
      const v16bf bv = cat16(*(const v8bf*)vr, *(const v8bf*)(vr + 8));
      if (half == 0)
        c0 = __builtin_amdgcn_wmma_f32_16x16x32_bf16(false, aa, false, bv,
                                                     (short)0, c0, false, false);
      else
        c1 = __builtin_amdgcn_wmma_f32_16x16x32_bf16(false, aa, false, bv,
                                                     (short)0, c1, false, false);
    }
  }

  // per-row group norm + gn scale/shift
#pragma unroll
  for (int i = 0; i < 8; ++i) {
    yst[(i + rowoff) * 32 + r16]      = c0[i];
    yst[(i + rowoff) * 32 + 16 + r16] = c1[i];
  }
  if (lane < 16) {
    float s = 0.0f, ss = 0.0f;
#pragma unroll
    for (int c = 0; c < HDIM; ++c) { const float v = yst[lane * 32 + c]; s += v; ss += v * v; }
    const float mu   = s * (1.0f / HDIM);
    const float var  = ss * (1.0f / HDIM) - mu * mu;
    const float rsig = rsqrtf(var + EPSF);
#pragma unroll
    for (int c = 0; c < HDIM; ++c)
      yst[lane * 32 + c] = (yst[lane * 32 + c] - mu) * rsig;
  }
  {
    const int row = r16;
    const int ch  = hb * 16;
#pragma unroll
    for (int c = 0; c < 16; ++c) {
      const int cg = ch + c;
      const float v = yst[row * 32 + cg];
      Y[(size_t)(b * SS + n0 + row) * DD + h * HDIM + cg] =
          v * gnw[h * HDIM + cg] + gnb[h * HDIM + cg];
    }
  }
}

// =====================================================================
// Host-side orchestration
// =====================================================================
extern "C" void kernel_launch(void* const* d_in, const int* in_sizes, int n_in,
                              void* d_out, int out_size, void* d_ws, size_t ws_size,
                              hipStream_t stream) {
  const float* X     = (const float*)d_in[0];
  const float* Wq    = (const float*)d_in[1];
  const float* Wk    = (const float*)d_in[2];
  const float* Wv    = (const float*)d_in[3];
  const float* Wg    = (const float*)d_in[4];
  const float* Wo    = (const float*)d_in[5];
  const float* gn_w  = (const float*)d_in[6];
  const float* gn_b  = (const float*)d_in[7];
  const float* ln1_w = (const float*)d_in[8];
  const float* ln1_b = (const float*)d_in[9];
  const float* ln2_w = (const float*)d_in[10];
  const float* ln2_b = (const float*)d_in[11];
  const float* w1    = (const float*)d_in[12];
  const float* b1    = (const float*)d_in[13];
  const float* w2    = (const float*)d_in[14];
  const float* b2    = (const float*)d_in[15];
  const float* lnf_w = (const float*)d_in[16];
  const float* lnf_b = (const float*)d_in[17];
  float* out = (float*)d_out;

  const size_t TD = (size_t)NTOK * DD;     // 2M floats
  float* ws   = (float*)d_ws;
  float* xn   = ws;
  float* q    = xn + TD;
  float* k    = q + TD;
  float* v    = k + TD;
  float* gbuf = v + TD;
  float* y    = gbuf + TD;
  float* yres = y + TD;
  float* f1   = q;                         // NTOK x 1024 == q..gbuf span
  __bf16* qbf = (__bf16*)(yres + TD);      // NTOK*DD bf16
  __bf16* kbf = qbf + TD;
  __bf16* vt  = kbf + TD;                  // (B,H,HD,S) bf16
  __bf16* wpk = vt + TD;                   // packed transposed weights

  const size_t WDD = (size_t)DD * DD;      // 65536
  const size_t WF  = (size_t)DD * FFND;    // 262144
  const size_t WLAYER = 5 * WDD + 2 * WF;  // per-layer packed elements

  const dim3 blk(256);
  const dim3 gProj(NTOK / 64, DD / 64);    // (128, 4)
  const dim3 gFfn1(NTOK / 64, FFND / 64);  // (128, 16)
  const dim3 gPackDD(DD / 64, DD / 64);    // (4, 4)
  const dim3 gPackW1(DD / 64, FFND / 64);  // (4, 16)
  const dim3 gPackW2(FFND / 64, DD / 64);  // (16, 4)

  // ---- pack all weights to bf16, transposed (every call; deterministic) ----
  for (int i = 0; i < LAYERS; ++i) {
    __bf16* wl = wpk + (size_t)i * WLAYER;
    wpack_kernel<<<gPackDD, blk, 0, stream>>>(Wq + i * WDD, wl + 0 * WDD, DD, DD);
    wpack_kernel<<<gPackDD, blk, 0, stream>>>(Wk + i * WDD, wl + 1 * WDD, DD, DD);
    wpack_kernel<<<gPackDD, blk, 0, stream>>>(Wv + i * WDD, wl + 2 * WDD, DD, DD);
    wpack_kernel<<<gPackDD, blk, 0, stream>>>(Wg + i * WDD, wl + 3 * WDD, DD, DD);
    wpack_kernel<<<gPackDD, blk, 0, stream>>>(Wo + i * WDD, wl + 4 * WDD, DD, DD);
    wpack_kernel<<<gPackW1, blk, 0, stream>>>(w1 + (size_t)i * WF, wl + 5 * WDD, DD, FFND);
    wpack_kernel<<<gPackW2, blk, 0, stream>>>(w2 + (size_t)i * WF, wl + 5 * WDD + WF, FFND, DD);
  }

  for (int i = 0; i < LAYERS; ++i) {
    const float* xin  = (i == 0) ? X : out;
    float*       xout = out;
    const __bf16* wl  = wpk + (size_t)i * WLAYER;
    const __bf16* WqT = wl + 0 * WDD;
    const __bf16* WkT = wl + 1 * WDD;
    const __bf16* WvT = wl + 2 * WDD;
    const __bf16* WgT = wl + 3 * WDD;
    const __bf16* WoT = wl + 4 * WDD;
    const __bf16* w1T = wl + 5 * WDD;
    const __bf16* w2T = wl + 5 * WDD + WF;

    ln_kernel<<<NTOK, blk, 0, stream>>>(xin, ln1_w + i * DD, ln1_b + i * DD, xn);
    gemm_bf16_kernel<<<gProj, blk, 0, stream>>>(xn, WqT, q,    NTOK, DD, DD, nullptr, nullptr, 0);
    gemm_bf16_kernel<<<gProj, blk, 0, stream>>>(xn, WkT, k,    NTOK, DD, DD, nullptr, nullptr, 0);
    gemm_bf16_kernel<<<gProj, blk, 0, stream>>>(xn, WvT, v,    NTOK, DD, DD, nullptr, nullptr, 0);
    gemm_bf16_kernel<<<gProj, blk, 0, stream>>>(xn, WgT, gbuf, NTOK, DD, DD, nullptr, nullptr, 0);
    xpos_kernel<<<(NTOK * (DD / 2)) / 256, blk, 0, stream>>>(q, k, qbf, kbf);
    vtpack_kernel<<<BB * HH * (SS / 64), blk, 0, stream>>>(v, vt);
    retention_kernel<<<(BB * HH * (SS / 16)) / 8, blk, 0, stream>>>(
        qbf, kbf, vt, gn_w + i * DD, gn_b + i * DD, y);
    gate_kernel<<<(int)(TD / 4 / 256), blk, 0, stream>>>(
        (const float4*)gbuf, (const float4*)y, (float4*)q);
    gemm_bf16_kernel<<<gProj, blk, 0, stream>>>(q, WoT, yres, NTOK, DD, DD, nullptr, xin, 4);
    ln_kernel<<<NTOK, blk, 0, stream>>>(yres, ln2_w + i * DD, ln2_b + i * DD, xn);
    gemm_bf16_kernel<<<gFfn1, blk, 0, stream>>>(xn, w1T, f1, NTOK, FFND, DD,
                                                b1 + i * FFND, nullptr, 1 | 2);
    gemm_bf16_kernel<<<gProj, blk, 0, stream>>>(f1, w2T, xn, NTOK, DD, FFND,
                                                b2 + i * DD, yres, 1 | 4);
    ln_kernel<<<NTOK, blk, 0, stream>>>(xn, lnf_w, lnf_b, xout);
  }
}
